// SparseMoeBlock_58480274702427
// MI455X (gfx1250) — matrix operand order repaired and checked by reading, not compiled, not run on hardware
//
#include <hip/hip_runtime.h>
#include <hip/hip_bf16.h>

typedef __bf16 bf16;
typedef __attribute__((ext_vector_type(16))) __bf16 v16bf;
typedef __attribute__((ext_vector_type(8)))  __bf16 v8bf;
typedef __attribute__((ext_vector_type(8)))  float  v8f;

constexpr int B_ = 2;
constexpr int S_ = 1024;
constexpr int T_ = B_ * S_;     // 2048 tokens
constexpr int H_ = 2048;
constexpr int ER = 8;           // routed experts
constexpr int E_ = 9;           // + uncond expert
constexpr int IM = 1408;        // moe intermediate
constexpr int ISH = 5632;       // shared intermediate
constexpr int UNCOND_LABEL = 1000;
constexpr float EPS_ = 1e-12f;

// ---------------------------------------------------------------------------
// Stage 1: normalize cluster centers (8 waves, one expert each)
// ---------------------------------------------------------------------------
__global__ void normalize_centers_kernel(const float* __restrict__ c,
                                         float* __restrict__ cn) {
    int wave = threadIdx.x >> 5;   // 0..7 == expert
    int lane = threadIdx.x & 31;
    const float* src = c + (size_t)wave * H_;
    float ss = 0.f;
    for (int h = lane; h < H_; h += 32) { float v = src[h]; ss += v * v; }
    #pragma unroll
    for (int off = 16; off > 0; off >>= 1) ss += __shfl_xor(ss, off, 32);
    float inv = rsqrtf(ss + EPS_);
    float* dst = cn + (size_t)wave * H_;
    for (int h = lane; h < H_; h += 32) dst[h] = src[h] * inv;
}

// ---------------------------------------------------------------------------
// Stage 2: router — one wave per token. cosine sim -> softmax -> top2 -> cw[T][E]
// ---------------------------------------------------------------------------
__global__ void router_kernel(const float* __restrict__ x,
                              const int* __restrict__ labels,
                              const float* __restrict__ cn,
                              float* __restrict__ cw) {
    int t = (int)((blockIdx.x * blockDim.x + threadIdx.x) >> 5);
    int lane = threadIdx.x & 31;
    if (t >= T_) return;
    const float* xr = x + (size_t)t * H_;
    float xx = 0.f;
    float dot[ER];
    #pragma unroll
    for (int e = 0; e < ER; ++e) dot[e] = 0.f;
    for (int h = lane; h < H_; h += 32) {
        float xv = xr[h];
        xx += xv * xv;
        #pragma unroll
        for (int e = 0; e < ER; ++e) dot[e] += xv * cn[(size_t)e * H_ + h];
    }
    #pragma unroll
    for (int off = 16; off > 0; off >>= 1) {
        xx += __shfl_xor(xx, off, 32);
        #pragma unroll
        for (int e = 0; e < ER; ++e) dot[e] += __shfl_xor(dot[e], off, 32);
    }
    if (lane == 0) {
        float inv = rsqrtf(xx + EPS_);
        float p[ER];
        float m = -1e30f;
        #pragma unroll
        for (int e = 0; e < ER; ++e) { p[e] = dot[e] * inv; m = fmaxf(m, p[e]); }
        float sum = 0.f;
        #pragma unroll
        for (int e = 0; e < ER; ++e) { p[e] = __expf(p[e] - m); sum += p[e]; }
        float is = 1.f / sum;
        #pragma unroll
        for (int e = 0; e < ER; ++e) p[e] *= is;
        int i1 = 0;
        #pragma unroll
        for (int e = 1; e < ER; ++e) if (p[e] > p[i1]) i1 = e;
        int i2 = (i1 == 0) ? 1 : 0;
        #pragma unroll
        for (int e = 0; e < ER; ++e) if (e != i1 && p[e] > p[i2]) i2 = e;
        bool unc = (labels[t / S_] == UNCOND_LABEL);
        float w[E_];
        #pragma unroll
        for (int e = 0; e < E_; ++e) w[e] = 0.f;
        if (unc) w[E_ - 1] = 1.f;
        else     { w[i1] += p[i1]; w[i2] += p[i2]; }
        #pragma unroll
        for (int e = 0; e < E_; ++e) cw[(size_t)t * E_ + e] = w[e];
    }
}

// ---------------------------------------------------------------------------
// Stage 3: cast activations f32 -> bf16
// ---------------------------------------------------------------------------
__global__ void cast_f32_bf16_kernel(const float* __restrict__ in,
                                     bf16* __restrict__ out, size_t n) {
    size_t i = (size_t)blockIdx.x * blockDim.x + threadIdx.x;
    if (i < n) out[i] = (bf16)in[i];
}

// ---------------------------------------------------------------------------
// Stage 4: batched transpose + cast: in [K,N] f32 -> out [N,K] bf16
// (dims are all multiples of 32; batch = blockIdx.z)
// ---------------------------------------------------------------------------
__global__ void transpose_cast_kernel(const float* __restrict__ in,
                                      bf16* __restrict__ out, int K, int N) {
    __shared__ float tile[32][33];
    size_t boff = (size_t)blockIdx.z * K * N;
    const float* src = in + boff;
    bf16* dst = out + boff;
    int tx = threadIdx.x & 31, ty = threadIdx.x >> 5;   // 32 x 8 threads
    int k0 = blockIdx.y * 32, n0 = blockIdx.x * 32;
    #pragma unroll
    for (int r = ty; r < 32; r += 8)
        tile[r][tx] = src[(size_t)(k0 + r) * N + (n0 + tx)];
    __syncthreads();
    #pragma unroll
    for (int r = ty; r < 32; r += 8)
        dst[(size_t)(n0 + r) * K + (k0 + tx)] = (bf16)tile[tx][r];
}

// ---------------------------------------------------------------------------
// WMMA fragment helpers (v_wmma_f32_16x16x32_bf16 VGPR layouts per CDNA5 ISA)
// ---------------------------------------------------------------------------
__device__ inline v16bf load_a_frag(const bf16* __restrict__ A, int lda, int lane) {
    // A points at tile origin (row 0, k 0); A row-major [16 x 32]
    int m  = lane & 15;
    int ks = lane >> 4;                 // K-half select
    const bf16* p = A + (size_t)m * lda + ks * 8;
    v8bf lo = *(const v8bf*)(p);        // K = base+0..7
    v8bf hi = *(const v8bf*)(p + 16);   // K = base+16..23
    v16bf r;
    #pragma unroll
    for (int i = 0; i < 8; ++i) { r[i] = lo[i]; r[i + 8] = hi[i]; }
    return r;
}

__device__ inline v16bf load_b_frag(const bf16* __restrict__ BT, int ldb, int lane) {
    // BT stored [N, K] row-major; fragment covers B[k0..k0+31, n0..n0+15]
    int n  = lane & 15;
    int kh = lane >> 4;                 // 0: K 0..7,16..23, 1: K 8..15,24..31
    const bf16* p = BT + (size_t)n * ldb + kh * 8;
    v8bf lo = *(const v8bf*)(p);
    v8bf hi = *(const v8bf*)(p + 16);
    v16bf r;
    #pragma unroll
    for (int i = 0; i < 8; ++i) { r[i] = lo[i]; r[i + 8] = hi[i]; }
    return r;
}

__device__ inline v8f wmma_bf16(v16bf a, v16bf b, v8f c) {
    return __builtin_amdgcn_wmma_f32_16x16x32_bf16(
        false, a, false, b, (short)0, c, false, false);
}

// ---------------------------------------------------------------------------
// Stage 5: fused gate/up GEMM + SiLU epilogue.
//   ACT[z][t, n] = silu(X @ GT[z]^T) * (X @ UT[z]^T)
// Block: 256 thr = 8 waves (4 M x 2 N), wave tile 32x32 -> block tile 128x64.
// ---------------------------------------------------------------------------
__global__ __launch_bounds__(256) void mlp_up_kernel(
    const bf16* __restrict__ X,    // [T, K]
    const bf16* __restrict__ GT,   // [z][N, K]
    const bf16* __restrict__ UT,   // [z][N, K]
    bf16* __restrict__ ACT,        // [z][T, N]
    int K, int N, size_t wStride, size_t aStride) {
    int lane = threadIdx.x & 31;
    int wave = threadIdx.x >> 5;
    int wm = wave & 3, wn = wave >> 2;
    int t0 = blockIdx.y * 128 + wm * 32;
    int n0 = blockIdx.x * 64 + wn * 32;
    const bf16* gt = GT + (size_t)blockIdx.z * wStride;
    const bf16* ut = UT + (size_t)blockIdx.z * wStride;
    bf16* act = ACT + (size_t)blockIdx.z * aStride;

    v8f vz = {0.f, 0.f, 0.f, 0.f, 0.f, 0.f, 0.f, 0.f};
    v8f gacc[2][2], uacc[2][2];
    #pragma unroll
    for (int i = 0; i < 2; ++i)
        #pragma unroll
        for (int j = 0; j < 2; ++j) { gacc[i][j] = vz; uacc[i][j] = vz; }

    for (int k = 0; k < K; k += 32) {
        const bf16* xa = X + (size_t)t0 * K + k;
        v16bf a0 = load_a_frag(xa, K, lane);
        v16bf a1 = load_a_frag(xa + (size_t)16 * K, K, lane);
        const bf16* gb = gt + (size_t)n0 * K + k;
        const bf16* ub = ut + (size_t)n0 * K + k;
        v16bf bg0 = load_b_frag(gb, K, lane);
        v16bf bg1 = load_b_frag(gb + (size_t)16 * K, K, lane);
        v16bf bu0 = load_b_frag(ub, K, lane);
        v16bf bu1 = load_b_frag(ub + (size_t)16 * K, K, lane);
        gacc[0][0] = wmma_bf16(a0, bg0, gacc[0][0]);
        gacc[0][1] = wmma_bf16(a0, bg1, gacc[0][1]);
        gacc[1][0] = wmma_bf16(a1, bg0, gacc[1][0]);
        gacc[1][1] = wmma_bf16(a1, bg1, gacc[1][1]);
        uacc[0][0] = wmma_bf16(a0, bu0, uacc[0][0]);
        uacc[0][1] = wmma_bf16(a0, bu1, uacc[0][1]);
        uacc[1][0] = wmma_bf16(a1, bu0, uacc[1][0]);
        uacc[1][1] = wmma_bf16(a1, bu1, uacc[1][1]);
    }

    int rbase = (lane >> 4) << 3;   // row offset 0 or 8 within 16x16 tile
    int cbase = lane & 15;
    #pragma unroll
    for (int mi = 0; mi < 2; ++mi)
        #pragma unroll
        for (int ni = 0; ni < 2; ++ni) {
            int row0 = t0 + mi * 16 + rbase;
            int col  = n0 + ni * 16 + cbase;
            #pragma unroll
            for (int r = 0; r < 8; ++r) {
                float g = gacc[mi][ni][r];
                float u = uacc[mi][ni][r];
                float a = (g / (1.f + __expf(-g))) * u;
                act[(size_t)(row0 + r) * N + col] = (bf16)a;
            }
        }
}

// ---------------------------------------------------------------------------
// Stage 6: down-proj + combine-weight accumulation over 9 experts + shared.
//   OUT[t,h] = sum_e cw[t,e] * (ACT_e[t,:] @ WdT_e[h,:]) + ACTS[t,:] @ WdsT[h,:]
// ---------------------------------------------------------------------------
__global__ __launch_bounds__(256) void moe_down_combine_kernel(
    const bf16* __restrict__ ACT,   // [E][T, IM]
    const bf16* __restrict__ WDT,   // [E][H, IM]
    const bf16* __restrict__ ACTS,  // [T, ISH]
    const bf16* __restrict__ WDST,  // [H, ISH]
    const float* __restrict__ CW,   // [T, E]
    float* __restrict__ OUT) {      // [T, H]
    __shared__ float cw_s[128][E_];
    int lane = threadIdx.x & 31;
    int wave = threadIdx.x >> 5;
    int wm = wave & 3, wn = wave >> 2;
    int tBlock = blockIdx.y * 128;
    int t0 = tBlock + wm * 32;
    int h0 = blockIdx.x * 64 + wn * 32;

    for (int idx = threadIdx.x; idx < 128 * E_; idx += 256) {
        int r = idx / E_, e = idx % E_;
        cw_s[r][e] = CW[(size_t)(tBlock + r) * E_ + e];
    }
    __syncthreads();

    v8f vz = {0.f, 0.f, 0.f, 0.f, 0.f, 0.f, 0.f, 0.f};
    v8f tacc[2][2];
    #pragma unroll
    for (int i = 0; i < 2; ++i)
        #pragma unroll
        for (int j = 0; j < 2; ++j) tacc[i][j] = vz;

    int rbase = (lane >> 4) << 3;

    for (int e = 0; e < E_; ++e) {
        const bf16* act = ACT + (size_t)e * T_ * IM;
        const bf16* wdt = WDT + (size_t)e * H_ * IM;
        v8f eacc[2][2];
        #pragma unroll
        for (int i = 0; i < 2; ++i)
            #pragma unroll
            for (int j = 0; j < 2; ++j) eacc[i][j] = vz;

        for (int k = 0; k < IM; k += 32) {
            const bf16* aa = act + (size_t)t0 * IM + k;
            v16bf a0 = load_a_frag(aa, IM, lane);
            v16bf a1 = load_a_frag(aa + (size_t)16 * IM, IM, lane);
            const bf16* bb = wdt + (size_t)h0 * IM + k;
            v16bf b0 = load_b_frag(bb, IM, lane);
            v16bf b1 = load_b_frag(bb + (size_t)16 * IM, IM, lane);
            eacc[0][0] = wmma_bf16(a0, b0, eacc[0][0]);
            eacc[0][1] = wmma_bf16(a0, b1, eacc[0][1]);
            eacc[1][0] = wmma_bf16(a1, b0, eacc[1][0]);
            eacc[1][1] = wmma_bf16(a1, b1, eacc[1][1]);
        }
        // per-token combine-weight scale, accumulate into total
        #pragma unroll
        for (int mi = 0; mi < 2; ++mi)
            #pragma unroll
            for (int ni = 0; ni < 2; ++ni)
                #pragma unroll
                for (int r = 0; r < 8; ++r) {
                    int rowL = wm * 32 + mi * 16 + rbase + r;
                    tacc[mi][ni][r] += cw_s[rowL][e] * eacc[mi][ni][r];
                }
    }

    // shared expert down-proj accumulates straight into total (weight 1.0)
    for (int k = 0; k < ISH; k += 32) {
        const bf16* aa = ACTS + (size_t)t0 * ISH + k;
        v16bf a0 = load_a_frag(aa, ISH, lane);
        v16bf a1 = load_a_frag(aa + (size_t)16 * ISH, ISH, lane);
        const bf16* bb = WDST + (size_t)h0 * ISH + k;
        v16bf b0 = load_b_frag(bb, ISH, lane);
        v16bf b1 = load_b_frag(bb + (size_t)16 * ISH, ISH, lane);
        tacc[0][0] = wmma_bf16(a0, b0, tacc[0][0]);
        tacc[0][1] = wmma_bf16(a0, b1, tacc[0][1]);
        tacc[1][0] = wmma_bf16(a1, b0, tacc[1][0]);
        tacc[1][1] = wmma_bf16(a1, b1, tacc[1][1]);
    }

    int cbase = lane & 15;
    #pragma unroll
    for (int mi = 0; mi < 2; ++mi)
        #pragma unroll
        for (int ni = 0; ni < 2; ++ni) {
            int row0 = t0 + mi * 16 + rbase;
            int col  = h0 + ni * 16 + cbase;
            #pragma unroll
            for (int r = 0; r < 8; ++r)
                OUT[(size_t)(row0 + r) * H_ + col] = tacc[mi][ni][r];
        }
}

// ---------------------------------------------------------------------------
// Host launcher
// ---------------------------------------------------------------------------
extern "C" void kernel_launch(void* const* d_in, const int* in_sizes, int n_in,
                              void* d_out, int out_size, void* d_ws, size_t ws_size,
                              hipStream_t stream) {
    const float* x       = (const float*)d_in[0];   // [B,S,H]
    const int*   labels  = (const int*)  d_in[1];   // [B]
    const float* centers = (const float*)d_in[2];   // [8,H]
    const float* Wg      = (const float*)d_in[3];   // [9,H,IM]
    const float* Wu      = (const float*)d_in[4];   // [9,H,IM]
    const float* Wd      = (const float*)d_in[5];   // [9,IM,H]
    const float* Wgs     = (const float*)d_in[6];   // [H,ISH]
    const float* Wus     = (const float*)d_in[7];   // [H,ISH]
    const float* Wds     = (const float*)d_in[8];   // [ISH,H]
    float* out = (float*)d_out;

    char* ws = (char*)d_ws;
    auto alloc = [&](size_t bytes) -> char* {
        char* p = ws;
        ws += (bytes + 255) & ~(size_t)255;
        return p;
    };
    bf16* xb   = (bf16*)alloc((size_t)T_ * H_ * 2);
    bf16* WgT  = (bf16*)alloc((size_t)E_ * IM * H_ * 2);    // [E][IM,H]
    bf16* WuT  = (bf16*)alloc((size_t)E_ * IM * H_ * 2);    // [E][IM,H]
    bf16* WdT  = (bf16*)alloc((size_t)E_ * H_ * IM * 2);    // [E][H,IM]
    bf16* WgsT = (bf16*)alloc((size_t)ISH * H_ * 2);        // [ISH,H]
    bf16* WusT = (bf16*)alloc((size_t)ISH * H_ * 2);        // [ISH,H]
    bf16* WdsT = (bf16*)alloc((size_t)H_ * ISH * 2);        // [H,ISH]
    bf16* ACT  = (bf16*)alloc((size_t)E_ * T_ * IM * 2);    // [E][T,IM]
    bf16* ACTS = (bf16*)alloc((size_t)T_ * ISH * 2);        // [T,ISH]
    float* cn  = (float*)alloc((size_t)ER * H_ * 4);
    float* cw  = (float*)alloc((size_t)T_ * E_ * 4);

    // 1) normalized centers (8 waves)
    normalize_centers_kernel<<<1, 256, 0, stream>>>(centers, cn);

    // 2) router: one wave per token, 8 waves / block
    router_kernel<<<T_ / 8, 256, 0, stream>>>(x, labels, cn, cw);

    // 3) cast x -> bf16
    {
        size_t n = (size_t)T_ * H_;
        cast_f32_bf16_kernel<<<(unsigned)((n + 255) / 256), 256, 0, stream>>>(x, xb, n);
    }

    // 4) weight transpose+cast: in [K,N] -> out [N,K] bf16
    transpose_cast_kernel<<<dim3(IM / 32, H_ / 32, E_), 256, 0, stream>>>(Wg, WgT, H_, IM);
    transpose_cast_kernel<<<dim3(IM / 32, H_ / 32, E_), 256, 0, stream>>>(Wu, WuT, H_, IM);
    transpose_cast_kernel<<<dim3(H_ / 32, IM / 32, E_), 256, 0, stream>>>(Wd, WdT, IM, H_);
    transpose_cast_kernel<<<dim3(ISH / 32, H_ / 32, 1), 256, 0, stream>>>(Wgs, WgsT, H_, ISH);
    transpose_cast_kernel<<<dim3(ISH / 32, H_ / 32, 1), 256, 0, stream>>>(Wus, WusT, H_, ISH);
    transpose_cast_kernel<<<dim3(H_ / 32, ISH / 32, 1), 256, 0, stream>>>(Wds, WdsT, ISH, H_);

    // 5) gate/up + SiLU: routed experts (z = 0..8) and shared expert
    mlp_up_kernel<<<dim3(IM / 64, T_ / 128, E_), 256, 0, stream>>>(
        xb, WgT, WuT, ACT, H_, IM, (size_t)IM * H_, (size_t)T_ * IM);
    mlp_up_kernel<<<dim3(ISH / 64, T_ / 128, 1), 256, 0, stream>>>(
        xb, WgsT, WusT, ACTS, H_, ISH, 0, 0);

    // 6) down-proj + combine + shared add -> fp32 out
    moe_down_combine_kernel<<<dim3(H_ / 64, T_ / 128), 256, 0, stream>>>(
        ACT, WdT, ACTS, WdsT, cw, out);
}
